// GIN_23665269801081
// MI455X (gfx1250) — matrix operand order repaired and checked by reading, not compiled, not run on hardware
//
#include <hip/hip_runtime.h>

// ---------------------------------------------------------------------------
// GIN layer for MI455X (gfx1250, wave32):
//   agg = x + segment_sum(x[src], dst)
//   h1  = relu(agg @ W1 + b1)            (bf16 WMMA, fp32 accum)
//   h2  = h1 @ W2 + b2                   (bf16 WMMA, fp32 accum)
//   out = relu(BN(h2)) @ Wc + bc
// ---------------------------------------------------------------------------

typedef __attribute__((ext_vector_type(16))) __bf16 v16bf;
typedef __attribute__((ext_vector_type(8)))  float  v8f;

union AFrag { v16bf v; uint4 q[2]; };
union BFrag { v16bf v; uint4 q[2]; };

__device__ __forceinline__ unsigned short f32_to_bf16(float f) {
    unsigned u = __float_as_uint(f);
    // round-to-nearest-even
    unsigned r = (u + 0x7FFFu + ((u >> 16) & 1u)) >> 16;
    return (unsigned short)r;
}

// ---------------------------------------------------------------- utilities
__global__ void gin_zero(float* __restrict__ p, int n) {
    int i = blockIdx.x * blockDim.x + threadIdx.x;
    if (i < n) p[i] = 0.0f;
}

// agg = x (float4 copy), then scatter adds on top
__global__ void gin_copy4(const float4* __restrict__ src, float4* __restrict__ dst, int n4) {
    int i = blockIdx.x * blockDim.x + threadIdx.x;
    if (i < n4) dst[i] = src[i];
}

// one edge handled by 16 threads, 4 features each (float4 load + 4 f32 atomics)
__global__ void gin_scatter(const float* __restrict__ x,
                            const int* __restrict__ src,
                            const int* __restrict__ dst,
                            float* __restrict__ agg, int E) {
    int t = blockIdx.x * blockDim.x + threadIdx.x;
    int e = t >> 4;
    if (e >= E) return;
    int q = (t & 15) * 4;
    int s = src[e], d = dst[e];
    const float4 v = *(const float4*)(x + (size_t)s * 64 + q);
    float* o = agg + (size_t)d * 64 + q;
    atomicAdd(o + 0, v.x);
    atomicAdd(o + 1, v.y);
    atomicAdd(o + 2, v.z);
    atomicAdd(o + 3, v.w);
}

__global__ void gin_cvt_bf16(const float* __restrict__ in, unsigned short* __restrict__ out, int n) {
    int i = blockIdx.x * blockDim.x + threadIdx.x;
    if (i < n) out[i] = f32_to_bf16(in[i]);
}

// Pack row-major W[K,N] (f32) into WMMA-B fragment-major bf16:
//   out[(((nTile*kSteps)+kStep)*32 + lane)*16 + j] = W[kStep*32 + (lane>>4)*16 + j, nTile*16 + (lane&15)]
// so each lane's 16 bf16 values (32 B) are contiguous.
__global__ void gin_pack_w(const float* __restrict__ W, unsigned short* __restrict__ out,
                           int K, int N) {
    int idx = blockIdx.x * blockDim.x + threadIdx.x;
    int kSteps = K >> 5;
    int total = kSteps * (N >> 4) * 32 * 16;
    if (idx >= total) return;
    int j    = idx & 15;
    int lane = (idx >> 4) & 31;
    int tile = idx >> 9;
    int nTile = tile / kSteps;
    int kStep = tile - nTile * kSteps;
    int n = nTile * 16 + (lane & 15);
    int k = kStep * 32 + ((lane >> 4) << 4) + j;
    out[idx] = f32_to_bf16(W[(size_t)k * N + n]);
}

// ------------------------------------------------------------------- GEMMs
// One wave -> 16(M) x 64(N) strip, 4 accumulators, K looped in steps of 32.
// OUT_N fixed at 512 (both GEMMs output 512 columns).
template <bool RELU_BF16_OUT>
__global__ __launch_bounds__(128)
void gin_gemm_wmma(const __bf16* __restrict__ A,            // row-major [M,K] bf16
                   const unsigned short* __restrict__ Bp,   // packed fragments
                   const float* __restrict__ bias,          // [512]
                   unsigned short* __restrict__ outBf,      // [M,512] bf16 (if RELU_BF16_OUT)
                   float* __restrict__ outF,                // [M,512] f32 (else)
                   int M, int K) {
    const int kSteps = K >> 5;
    const int wave = threadIdx.x >> 5;
    const int lane = threadIdx.x & 31;
    const int h    = lane >> 4;          // half-wave
    const int mrow = lane & 15;

    const int mTile = blockIdx.x * 4 + wave;
    const int mTiles = (M + 15) >> 4;
    if (mTile >= mTiles) return;         // wave-uniform: EXEC stays all-ones
    const int m0 = mTile << 4;
    const int nBase = blockIdx.y << 6;   // 64-column strip
    const int nTileBase = nBase >> 4;

    int arow = m0 + mrow;
    if (arow >= M) arow = M - 1;         // clamp: keep EXEC full, discard at store
    const __bf16* Arow = A + (size_t)arow * K;

    v8f acc[4] = {};

    for (int ks = 0; ks < kSteps; ++ks) {
        AFrag a;
        const uint4* ap = (const uint4*)(Arow + (ks << 5) + (h << 4));
        a.q[0] = ap[0];                  // K = ks*32 + 16h + 0..7
        a.q[1] = ap[1];                  // K = ks*32 + 16h + 8..15
#pragma unroll
        for (int t = 0; t < 4; ++t) {
            BFrag b;
            const uint4* bp = (const uint4*)(Bp +
                ((((size_t)(nTileBase + t) * kSteps + ks) * 32 + lane) << 4));
            b.q[0] = bp[0];
            b.q[1] = bp[1];
            acc[t] = __builtin_amdgcn_wmma_f32_16x16x32_bf16(
                /*neg_a=*/false, a.v, /*neg_b=*/false, b.v,
                /*c_mod=*/(short)0, acc[t], /*reuse_a=*/false, /*reuse_b=*/false);
        }
    }

    // Epilogue: D layout -> row = m0 + r + 8h, col = nBase + t*16 + (lane&15)
    const int colLane = nBase + (lane & 15);
#pragma unroll
    for (int t = 0; t < 4; ++t) {
        const int col = colLane + (t << 4);
        const float bv = bias[col];
#pragma unroll
        for (int r = 0; r < 8; ++r) {
            const int row = m0 + r + (h << 3);
            if (row < M) {
                float v = acc[t][r] + bv;
                if (RELU_BF16_OUT) {
                    outBf[(size_t)row * 512 + col] = f32_to_bf16(fmaxf(v, 0.0f));
                } else {
                    outF[(size_t)row * 512 + col] = v;
                }
            }
        }
    }
}

// ------------------------------------------------------------ batchnorm + cls
__global__ void gin_bn_stats(const float* __restrict__ h2, float* __restrict__ stats, int M) {
    int f  = blockIdx.x * blockDim.x + threadIdx.x;   // 0..511 (coalesced across lanes)
    if (f >= 512) return;
    int r0 = blockIdx.y * 512;
    int r1 = r0 + 512; if (r1 > M) r1 = M;
    float s = 0.0f, s2 = 0.0f;
    for (int r = r0; r < r1; ++r) {
        float v = h2[(size_t)r * 512 + f];
        s  += v;
        s2 += v * v;
    }
    atomicAdd(&stats[f], s);
    atomicAdd(&stats[512 + f], s2);
}

__global__ void gin_bn_finalize(const float* __restrict__ stats,
                                const float* __restrict__ gamma,
                                const float* __restrict__ beta,
                                float* __restrict__ ss, int M) {
    int f = blockIdx.x * blockDim.x + threadIdx.x;
    if (f >= 512) return;
    float inv  = 1.0f / (float)M;
    float mean = stats[f] * inv;
    float var  = stats[512 + f] * inv - mean * mean;   // biased (jnp.var)
    float sc   = gamma[f] * rsqrtf(var + 1e-5f);
    ss[f]       = sc;
    ss[512 + f] = beta[f] - mean * sc;
}

// one wave per row: normalize + relu + 512->2 dot, wave32 shuffle reduction
__global__ __launch_bounds__(256)
void gin_bn_cls(const float* __restrict__ h2, const float* __restrict__ ss,
                const float* __restrict__ Wc, const float* __restrict__ bc,
                float* __restrict__ out, int M) {
    int row  = (blockIdx.x * blockDim.x + threadIdx.x) >> 5;
    int lane = threadIdx.x & 31;
    if (row >= M) return;
    const float* hrow = h2 + (size_t)row * 512;
    float a0 = 0.0f, a1 = 0.0f;
    int f0 = lane << 4;                  // 16 features per lane
#pragma unroll
    for (int j = 0; j < 16; ++j) {
        int f = f0 + j;
        float v = fmaxf(hrow[f] * ss[f] + ss[512 + f], 0.0f);
        a0 += v * Wc[2 * f + 0];
        a1 += v * Wc[2 * f + 1];
    }
#pragma unroll
    for (int off = 16; off > 0; off >>= 1) {
        a0 += __shfl_down(a0, off, 32);
        a1 += __shfl_down(a1, off, 32);
    }
    if (lane == 0) {
        out[2 * row + 0] = a0 + bc[0];
        out[2 * row + 1] = a1 + bc[1];
    }
}

// ---------------------------------------------------------------------------
static inline size_t al256(size_t x) { return (x + 255) & ~(size_t)255; }

extern "C" void kernel_launch(void* const* d_in, const int* in_sizes, int n_in,
                              void* d_out, int out_size, void* d_ws, size_t ws_size,
                              hipStream_t stream) {
    const float* x     = (const float*)d_in[0];
    const int*   edges = (const int*)d_in[1];
    const float* W1    = (const float*)d_in[2];
    const float* b1    = (const float*)d_in[3];
    const float* W2    = (const float*)d_in[4];
    const float* b2    = (const float*)d_in[5];
    const float* gamma = (const float*)d_in[6];
    const float* beta  = (const float*)d_in[7];
    const float* Wc    = (const float*)d_in[8];
    const float* bc    = (const float*)d_in[9];
    float* out = (float*)d_out;

    const int M = in_sizes[0] / 64;       // 50000 nodes
    const int E = in_sizes[1] / 2;        // 800000 edges
    const int F = 64, H = 512;

    // ------- workspace layout -------
    char* ws = (char*)d_ws;
    size_t off = 0;
    float*          agg  = (float*)(ws + off);          off = al256(off + (size_t)M * F * 4);
    unsigned short* h0b  = (unsigned short*)(ws + off); off = al256(off + (size_t)M * F * 2);
    unsigned short* w1p  = (unsigned short*)(ws + off); off = al256(off + (size_t)F * H * 2);
    unsigned short* w2p  = (unsigned short*)(ws + off); off = al256(off + (size_t)H * H * 2);
    unsigned short* h1b  = (unsigned short*)(ws + off); off = al256(off + (size_t)M * H * 2);
    float*          h2   = (float*)(ws + off);          off = al256(off + (size_t)M * H * 4);
    float*          stats= (float*)(ws + off);          off = al256(off + 1024 * 4);
    float*          ssbuf= (float*)(ws + off);          off = al256(off + 1024 * 4);
    (void)ws_size; (void)n_in; (void)out_size;

    const int src_off = 0, dst_off = E;

    // 1) zero BN accumulators
    gin_zero<<<(1024 + 255) / 256, 256, 0, stream>>>(stats, 1024);

    // 2) agg = x, then scatter-add x[src] into agg[dst]
    {
        int n4 = M * F / 4;
        gin_copy4<<<(n4 + 255) / 256, 256, 0, stream>>>((const float4*)x, (float4*)agg, n4);
    }
    gin_scatter<<<((size_t)E * 16 + 255) / 256, 256, 0, stream>>>(
        x, edges + src_off, edges + dst_off, agg, E);

    // 3) h0 -> bf16
    gin_cvt_bf16<<<((size_t)M * F + 255) / 256, 256, 0, stream>>>(agg, h0b, M * F);

    // 4) pack weights into WMMA-B fragment layout
    gin_pack_w<<<((F / 32) * (H / 16) * 512 + 255) / 256, 256, 0, stream>>>(W1, w1p, F, H);
    gin_pack_w<<<((H / 32) * (H / 16) * 512 + 255) / 256, 256, 0, stream>>>(W2, w2p, H, H);

    // 5) h1 = relu(h0 @ W1 + b1)  -> bf16
    {
        dim3 grid((((M + 15) / 16) + 3) / 4, H / 64);
        gin_gemm_wmma<true><<<grid, 128, 0, stream>>>(
            (const __bf16*)h0b, w1p, b1, h1b, nullptr, M, F);
    }
    // 6) h2 = h1 @ W2 + b2  -> f32
    {
        dim3 grid((((M + 15) / 16) + 3) / 4, H / 64);
        gin_gemm_wmma<false><<<grid, 128, 0, stream>>>(
            (const __bf16*)h1b, w2p, b2, nullptr, h2, M, H);
    }

    // 7) BN statistics (sum / sumsq per feature)
    {
        dim3 grid(2, (M + 511) / 512);
        gin_bn_stats<<<grid, 256, 0, stream>>>(h2, stats, M);
    }
    gin_bn_finalize<<<2, 256, 0, stream>>>(stats, gamma, beta, ssbuf, M);

    // 8) normalize + relu + classifier
    gin_bn_cls<<<((size_t)M * 32 + 255) / 256, 256, 0, stream>>>(h2, ssbuf, Wc, bc, out, M);
}
